// GraphDegreeConv_76270029242554
// MI455X (gfx1250) — matrix operand order repaired and checked by reading, not compiled, not run on hardware
//
#include <hip/hip_runtime.h>
#include <hip/hip_bf16.h>

// ---------------------------------------------------------------------------
// GraphDegreeConv fused WMMA kernel for gfx1250 (MI455X)
//   stage 1: pack combined weights [Wself | Wdeg] -> bf16, pre-swizzled into
//            per-lane WMMA B-fragment order (contiguous 32B/lane loads)
//   stage 2: fused gather + sum + concat + GEMM (v_wmma_f32_16x16x32_bf16),
//            A tile staged in LDS as bf16 (converted once, not per wave)
//   stage 3: deterministic BatchNorm (biased) + ReLU, in place on d_out
// ---------------------------------------------------------------------------

typedef __attribute__((ext_vector_type(16))) __bf16 v16bf;
typedef __attribute__((ext_vector_type(8)))  __bf16 v8bf;
typedef __attribute__((ext_vector_type(8)))  float  v8f;

#define N_NODES    200000
#define N_EDGES    600000
#define NODE_SIZE  256
#define EDGE_SIZE  64
#define OUT_SIZE   256
#define N_BUCKET   40000
#define K_TOTAL    576            // 256 self + 256 nbr-node + 64 nbr-edge
#define KSTEPS     18             // 576 / 32
#define M_TILE     32             // rows per block (40000 % 32 == 0)
#define LDSW       584            // padded LDS row stride (bf16 elems)
#define NPACK      (5 * KSTEPS * 16 * 32 * 16)   // bf16 elems of packed W
#define RED_BLOCKS 400
#define ROWS_PER_RB 500           // 400*500 = 200000

struct NbrPtrs { const int* node[5]; const int* edge[5]; };

static __device__ __forceinline__ unsigned short bfbits(float f) {
    // round-to-nearest-even f32 -> bf16 bits (inputs finite)
    unsigned u = __builtin_bit_cast(unsigned, f);
    return (unsigned short)((u + 0x7fffu + ((u >> 16) & 1u)) >> 16);
}

static __device__ __forceinline__ __bf16 f2bf(float f) {
    unsigned short h = bfbits(f);
    return __builtin_bit_cast(__bf16, h);
}

// pack 4 floats -> 4 bf16 (8 bytes) and store to LDS (8B aligned)
static __device__ __forceinline__ void store_bf4(__bf16* p, float4 v) {
    uint2 u;
    u.x = (unsigned)bfbits(v.x) | ((unsigned)bfbits(v.y) << 16);
    u.y = (unsigned)bfbits(v.z) | ((unsigned)bfbits(v.w) << 16);
    *(uint2*)p = u;
}

// A-fragment load: lane holds K = kb..kb+7 (16B) and kb+16..kb+23 (16B)
static __device__ __forceinline__ v16bf load_afrag(const __bf16* p) {
    v8bf lo = *(const v8bf*)(p);
    v8bf hi = *(const v8bf*)(p + 16);
    return __builtin_shufflevector(lo, hi, 0, 1, 2, 3, 4, 5, 6, 7,
                                   8, 9, 10, 11, 12, 13, 14, 15);
}

// ---------------------------------------------------------------------------
// Pack W: wpack[(((b*18+s)*16+t)*32+lane)*16+j] = Wc[b][k][n] in bf16 with
//   n = t*16 + (lane&15),  k = s*32 + (lane>=16 ? 8 : 0) + (j<8 ? j : j+8)
// matching the 16-bit B-matrix (32x16, K-major per lane) WMMA layout.
// Wc[b][k<256][n] = W_self[n][k]; Wc[b][k>=256][n] = W_deg[b][n][k-256].
// ---------------------------------------------------------------------------
__global__ __launch_bounds__(256)
void pack_weights(const float* __restrict__ Wself,
                  const float* __restrict__ Wdeg,
                  __bf16* __restrict__ wpack) {
    size_t i = (size_t)blockIdx.x * blockDim.x + threadIdx.x;
    if (i >= (size_t)NPACK) return;
    int j    = (int)(i & 15);
    int lane = (int)((i >> 4) & 31);
    int t    = (int)((i >> 9) & 15);
    int bs   = (int)(i >> 13);          // b*18 + s
    int s    = bs % KSTEPS;
    int b    = bs / KSTEPS;
    int n = t * 16 + (lane & 15);
    int k = s * 32 + ((lane >> 4) ? 8 : 0) + (j < 8 ? j : j + 8);
    float v;
    if (k < NODE_SIZE) {
        v = Wself[(size_t)n * NODE_SIZE + k];
    } else {
        v = Wdeg[((size_t)b * OUT_SIZE + n) * (NODE_SIZE + EDGE_SIZE) + (k - NODE_SIZE)];
    }
    wpack[i] = f2bf(v);
}

// ---------------------------------------------------------------------------
// Fused gather + GEMM.  One block = 32 rows of one degree bucket.
// 256 threads (8 waves).  Wave w computes output columns [32w, 32w+32)
// for all 32 rows: 4 accumulators (2 M-tiles x 2 N-tiles).
// ---------------------------------------------------------------------------
__global__ __launch_bounds__(256)
void gdc_gemm(const float* __restrict__ node_repr,
              const float* __restrict__ edge_repr,
              NbrPtrs nbrs,
              const __bf16* __restrict__ wpack,
              const float* __restrict__ bias,
              float* __restrict__ out) {
    __shared__ __bf16 Xs[M_TILE * LDSW];        // 32 x 576 bf16 (pad 584)

    const int tid    = threadIdx.x;
    const int row0   = blockIdx.x * M_TILE;
    const int bucket = row0 / N_BUCKET;          // uniform per block
    const int deg    = bucket + 1;
    const int rl16   = tid >> 4;                 // 0..15
    const int c16    = tid & 15;                 // 0..15

    // ---- gather phase (coalesced float4 reads, bf16 packed LDS writes) ----
    #pragma unroll
    for (int rb = 0; rb < M_TILE; rb += 16) {
        const int rl = rb + rl16;
        const int r  = row0 + rl;
        const int li = r - bucket * N_BUCKET;
        const int* __restrict__ nnb = nbrs.node[bucket] + (size_t)li * deg;
        const int* __restrict__ enb = nbrs.edge[bucket] + (size_t)li * deg;
        __bf16* xrow = &Xs[rl * LDSW];

        // self features -> X[rl][0:256]
        #pragma unroll
        for (int p = 0; p < 4; ++p) {
            int col = p * 64 + c16 * 4;
            float4 v = *(const float4*)(node_repr + (size_t)r * NODE_SIZE + col);
            store_bf4(xrow + col, v);
        }
        // neighbor-node sum -> X[rl][256:512]
        #pragma unroll
        for (int p = 0; p < 4; ++p) {
            int col = p * 64 + c16 * 4;
            float4 a; a.x = a.y = a.z = a.w = 0.f;
            for (int j = 0; j < deg; ++j) {
                float4 v = *(const float4*)(node_repr + (size_t)nnb[j] * NODE_SIZE + col);
                a.x += v.x; a.y += v.y; a.z += v.z; a.w += v.w;
            }
            store_bf4(xrow + NODE_SIZE + col, a);
        }
        // neighbor-edge sum -> X[rl][512:576]
        {
            int col = c16 * 4;
            float4 a; a.x = a.y = a.z = a.w = 0.f;
            for (int j = 0; j < deg; ++j) {
                float4 v = *(const float4*)(edge_repr + (size_t)enb[j] * EDGE_SIZE + col);
                a.x += v.x; a.y += v.y; a.z += v.z; a.w += v.w;
            }
            store_bf4(xrow + 2 * NODE_SIZE + col, a);
        }
    }
    __syncthreads();

    // ---- WMMA phase ----
    const int wave = tid >> 5;
    const int lane = tid & 31;
    const int half = lane >> 4;                  // 0 or 1
    const int kb_a = half * 8;                   // A K-base within a 32-step

    v8f acc00 = {};   // rows 0-15,  cols n0..n0+15
    v8f acc01 = {};   // rows 0-15,  cols n0+16..n0+31
    v8f acc10 = {};   // rows 16-31, cols n0..n0+15
    v8f acc11 = {};   // rows 16-31, cols n0+16..n0+31

    const __bf16* __restrict__ xa0 = &Xs[(lane & 15) * LDSW + kb_a];
    const __bf16* __restrict__ xa1 = &Xs[((lane & 15) + 16) * LDSW + kb_a];
    const __bf16* __restrict__ wb =
        wpack + (size_t)bucket * (KSTEPS * 16 * 32 * 16)
              + ((size_t)2 * wave * 32 + lane) * 16;

    #pragma unroll 2
    for (int s = 0; s < KSTEPS; ++s) {
        const int k0 = s * 32;
        // A fragments (16x32 bf16) straight from LDS, no conversion
        v16bf a0 = load_afrag(xa0 + k0);
        v16bf a1 = load_afrag(xa1 + k0);
        // B fragments: pre-swizzled, 32 contiguous bytes per lane
        const __bf16* bbase = wb + (size_t)s * (16 * 32 * 16);
        v16bf b0 = *(const v16bf*)(bbase);
        v16bf b1 = *(const v16bf*)(bbase + 32 * 16);
        acc00 = __builtin_amdgcn_wmma_f32_16x16x32_bf16(
                    false, a0, false, b0, (short)0, acc00, false, false);
        acc01 = __builtin_amdgcn_wmma_f32_16x16x32_bf16(
                    false, a0, false, b1, (short)0, acc01, false, false);
        acc10 = __builtin_amdgcn_wmma_f32_16x16x32_bf16(
                    false, a1, false, b0, (short)0, acc10, false, false);
        acc11 = __builtin_amdgcn_wmma_f32_16x16x32_bf16(
                    false, a1, false, b1, (short)0, acc11, false, false);
    }

    // ---- epilogue: + bias, store pre-BN acts ----
    // C/D layout: lane (&15) = N, VGPR row rr -> M = rr + (lane>=16 ? 8 : 0)
    const int nc = wave * 32 + (lane & 15);
    const float bia0 = bias[nc];
    const float bia1 = bias[nc + 16];
    #pragma unroll
    for (int rr = 0; rr < 8; ++rr) {
        const int mrow = rr + half * 8;
        const size_t o0 = (size_t)(row0 + mrow) * OUT_SIZE;
        const size_t o1 = (size_t)(row0 + 16 + mrow) * OUT_SIZE;
        out[o0 + nc]      = acc00[rr] + bia0;
        out[o0 + nc + 16] = acc01[rr] + bia1;
        out[o1 + nc]      = acc10[rr] + bia0;
        out[o1 + nc + 16] = acc11[rr] + bia1;
    }
}

// ---------------------------------------------------------------------------
// Deterministic BatchNorm (biased stats) + ReLU over d_out columns.
// ---------------------------------------------------------------------------
__global__ __launch_bounds__(256)
void bn_partial(const float* __restrict__ out,
                float* __restrict__ psum, float* __restrict__ psum2) {
    const int c = threadIdx.x;          // column 0..255
    const int b = blockIdx.x;           // row chunk
    float s = 0.f, s2 = 0.f;
    size_t base = (size_t)b * ROWS_PER_RB * OUT_SIZE + c;
    for (int r = 0; r < ROWS_PER_RB; ++r) {
        float v = out[base + (size_t)r * OUT_SIZE];
        s += v; s2 += v * v;
    }
    psum [b * OUT_SIZE + c] = s;
    psum2[b * OUT_SIZE + c] = s2;
}

__global__ __launch_bounds__(256)
void bn_finalize(const float* __restrict__ psum, const float* __restrict__ psum2,
                 float* __restrict__ meanstd) {
    const int c = threadIdx.x;
    float s = 0.f, s2 = 0.f;
    for (int b = 0; b < RED_BLOCKS; ++b) {
        s  += psum [b * OUT_SIZE + c];
        s2 += psum2[b * OUT_SIZE + c];
    }
    const float inv_n = 1.0f / (float)N_NODES;
    float mean = s * inv_n;
    float var  = s2 * inv_n - mean * mean;
    meanstd[c]            = mean;
    meanstd[OUT_SIZE + c] = rsqrtf(var + 1e-5f);
}

__global__ __launch_bounds__(256)
void bn_apply(float* __restrict__ out, const float* __restrict__ meanstd) {
    __shared__ float mu[OUT_SIZE], is[OUT_SIZE];
    const int c = threadIdx.x;
    mu[c] = meanstd[c];
    is[c] = meanstd[OUT_SIZE + c];
    __syncthreads();
    for (size_t r = blockIdx.x; r < (size_t)N_NODES; r += gridDim.x) {
        size_t o = r * OUT_SIZE + c;
        float v = (out[o] - mu[c]) * is[c];
        out[o] = v > 0.f ? v : 0.f;
    }
}

// ---------------------------------------------------------------------------
extern "C" void kernel_launch(void* const* d_in, const int* in_sizes, int n_in,
                              void* d_out, int out_size, void* d_ws, size_t ws_size,
                              hipStream_t stream) {
    (void)in_sizes; (void)n_in; (void)out_size; (void)ws_size;

    const float* node_repr = (const float*)d_in[0];
    const float* edge_repr = (const float*)d_in[1];
    NbrPtrs nbrs;
    // setup_inputs dict order interleaves node_nbr_d, edge_nbr_d
    for (int d = 0; d < 5; ++d) {
        nbrs.node[d] = (const int*)d_in[2 + 2 * d];
        nbrs.edge[d] = (const int*)d_in[3 + 2 * d];
    }
    const float* Wself = (const float*)d_in[12];
    const float* Wdeg  = (const float*)d_in[13];
    const float* bias  = (const float*)d_in[14];
    float* out = (float*)d_out;

    // workspace layout
    char* ws = (char*)d_ws;
    __bf16* wpack  = (__bf16*)ws;                                   // 1,474,560 B
    float*  psum   = (float*)(ws + (size_t)NPACK * 2);              //   409,600 B
    float*  psum2  = psum  + RED_BLOCKS * OUT_SIZE;                 //   409,600 B
    float*  meanstd= psum2 + RED_BLOCKS * OUT_SIZE;                 //     2,048 B

    // 1) pack weights to bf16 in WMMA B-fragment order
    pack_weights<<<(NPACK + 255) / 256, 256, 0, stream>>>(Wself, Wdeg, wpack);

    // 2) fused gather + GEMM (+bias), pre-BN acts into d_out
    gdc_gemm<<<N_NODES / M_TILE, 256, 0, stream>>>(
        node_repr, edge_repr, nbrs, wpack, bias, out);

    // 3) deterministic BatchNorm + ReLU in place
    bn_partial<<<RED_BLOCKS, 256, 0, stream>>>(out, psum, psum2);
    bn_finalize<<<1, 256, 0, stream>>>(psum, psum2, meanstd);
    bn_apply<<<2048, 256, 0, stream>>>(out, meanstd);
}
